// DepthNet_39900246179980
// MI455X (gfx1250) — matrix operand order repaired and checked by reading, not compiled, not run on hardware
//
#include <hip/hip_runtime.h>
#include <math.h>

#define Bn 2
#define Vn 5
#define Cn 32
#define Hn 128
#define Wn 160
#define Dn 32
#define HWn (Hn*Wn)

typedef __attribute__((ext_vector_type(16))) _Float16 v16h;
typedef __attribute__((ext_vector_type(8)))  float    v8f;

// ---------------- transpose (V,B,C,H,W) -> (V,B,H,W,C) : contiguous channel vectors for gathers
__global__ void k_transpose(const float* __restrict__ in, float* __restrict__ out) {
  int o = blockIdx.x * 256 + threadIdx.x;
  const int N = Vn * Bn * Cn * HWn;
  if (o >= N) return;
  int c = o & 31;
  int t = o >> 5;
  int w = t % Wn; t /= Wn;
  int h = t % Hn; t /= Hn;
  int b = t % Bn; t /= Bn;
  int v = t;
  out[o] = in[(((v * Bn + b) * Cn + c) * Hn + h) * Wn + w];
}

// ---------------- 4x4 helpers (row-major; adjugate inverse works row-major too)
__device__ inline void mat4inv(const float* m, float* invOut) {
  float inv[16];
  inv[0]  =  m[5]*m[10]*m[15] - m[5]*m[11]*m[14] - m[9]*m[6]*m[15] + m[9]*m[7]*m[14] + m[13]*m[6]*m[11] - m[13]*m[7]*m[10];
  inv[4]  = -m[4]*m[10]*m[15] + m[4]*m[11]*m[14] + m[8]*m[6]*m[15] - m[8]*m[7]*m[14] - m[12]*m[6]*m[11] + m[12]*m[7]*m[10];
  inv[8]  =  m[4]*m[9]*m[15]  - m[4]*m[11]*m[13] - m[8]*m[5]*m[15] + m[8]*m[7]*m[13] + m[12]*m[5]*m[11] - m[12]*m[7]*m[9];
  inv[12] = -m[4]*m[9]*m[14]  + m[4]*m[10]*m[13] + m[8]*m[5]*m[14] - m[8]*m[6]*m[13] - m[12]*m[5]*m[10] + m[12]*m[6]*m[9];
  inv[1]  = -m[1]*m[10]*m[15] + m[1]*m[11]*m[14] + m[9]*m[2]*m[15] - m[9]*m[3]*m[14] - m[13]*m[2]*m[11] + m[13]*m[3]*m[10];
  inv[5]  =  m[0]*m[10]*m[15] - m[0]*m[11]*m[14] - m[8]*m[2]*m[15] + m[8]*m[3]*m[14] + m[12]*m[2]*m[11] - m[12]*m[3]*m[10];
  inv[9]  = -m[0]*m[9]*m[15]  + m[0]*m[11]*m[13] + m[8]*m[1]*m[15] - m[8]*m[3]*m[13] - m[12]*m[1]*m[11] + m[12]*m[3]*m[9];
  inv[13] =  m[0]*m[9]*m[14]  - m[0]*m[10]*m[13] - m[8]*m[1]*m[14] + m[8]*m[2]*m[13] + m[12]*m[1]*m[10] - m[12]*m[2]*m[9];
  inv[2]  =  m[1]*m[6]*m[15]  - m[1]*m[7]*m[14]  - m[5]*m[2]*m[15] + m[5]*m[3]*m[14] + m[13]*m[2]*m[7]  - m[13]*m[3]*m[6];
  inv[6]  = -m[0]*m[6]*m[15]  + m[0]*m[7]*m[14]  + m[4]*m[2]*m[15] - m[4]*m[3]*m[14] - m[12]*m[2]*m[7]  + m[12]*m[3]*m[6];
  inv[10] =  m[0]*m[5]*m[15]  - m[0]*m[7]*m[13]  - m[4]*m[1]*m[15] + m[4]*m[3]*m[13] + m[12]*m[1]*m[7]  - m[12]*m[3]*m[5];
  inv[14] = -m[0]*m[5]*m[14]  + m[0]*m[6]*m[13]  + m[4]*m[1]*m[14] - m[4]*m[2]*m[13] - m[12]*m[1]*m[6]  + m[12]*m[2]*m[5];
  inv[3]  = -m[1]*m[6]*m[11]  + m[1]*m[7]*m[10]  + m[5]*m[2]*m[11] - m[5]*m[3]*m[10] - m[9]*m[2]*m[7]   + m[9]*m[3]*m[6];
  inv[7]  =  m[0]*m[6]*m[11]  - m[0]*m[7]*m[10]  - m[4]*m[2]*m[11] + m[4]*m[3]*m[10] + m[8]*m[2]*m[7]   - m[8]*m[3]*m[6];
  inv[11] = -m[0]*m[5]*m[11]  + m[0]*m[7]*m[9]   + m[4]*m[1]*m[11] - m[4]*m[3]*m[9]  - m[8]*m[1]*m[7]   + m[8]*m[3]*m[5];
  inv[15] =  m[0]*m[5]*m[10]  - m[0]*m[6]*m[9]   - m[4]*m[1]*m[10] + m[4]*m[2]*m[9]  + m[8]*m[1]*m[6]   - m[8]*m[2]*m[5];
  float det = m[0]*inv[0] + m[1]*inv[4] + m[2]*inv[8] + m[3]*inv[12];
  det = 1.0f / det;
  for (int i = 0; i < 16; ++i) invOut[i] = inv[i] * det;
}

__device__ inline void build_new(const float* p0, const float* p1, float* N) {
  // top3 = p1[:3,:3] @ p0[:3,:4]; row3 = p0[3,:]
  for (int r = 0; r < 3; ++r)
    for (int c = 0; c < 4; ++c)
      N[r*4+c] = p1[r*4+0]*p0[0*4+c] + p1[r*4+1]*p0[1*4+c] + p1[r*4+2]*p0[2*4+c];
  for (int c = 0; c < 4; ++c) N[12+c] = p0[12+c];
}

// projws[((v-1)*Bn+b)*12] = {rot 3x3 row-major, trans 3}
__global__ void k_proj(const float* __restrict__ pm, float* __restrict__ projws) {
  if (threadIdx.x != 0 || blockIdx.x != 0) return;
  for (int b = 0; b < Bn; ++b) {
    float R[16], Ri[16];
    const float* p0 = pm + ((b * Vn + 0) * 2 + 0) * 16;
    const float* p1 = pm + ((b * Vn + 0) * 2 + 1) * 16;
    build_new(p0, p1, R);
    mat4inv(R, Ri);
    for (int v = 1; v < Vn; ++v) {
      const float* q0 = pm + ((b * Vn + v) * 2 + 0) * 16;
      const float* q1 = pm + ((b * Vn + v) * 2 + 1) * 16;
      float S[16], P[16];
      build_new(q0, q1, S);
      for (int r = 0; r < 4; ++r)
        for (int c = 0; c < 4; ++c) {
          float a = 0.f;
          for (int k = 0; k < 4; ++k) a += S[r*4+k] * Ri[k*4+c];
          P[r*4+c] = a;
        }
      float* dst = projws + ((v - 1) * Bn + b) * 12;
      for (int r = 0; r < 3; ++r)
        for (int c = 0; c < 3; ++c) dst[r*3+c] = P[r*4+c];
      for (int r = 0; r < 3; ++r) dst[9+r] = P[r*4+3];
    }
  }
}

__global__ void k_init(float* __restrict__ sim_sum, float* __restrict__ w_sum) {
  int i = blockIdx.x * 256 + threadIdx.x;
  if (i < Bn * Dn * HWn) sim_sum[i] = 0.f;
  if (i < Bn * HWn)      w_sum[i]   = 1e-5f;
}

// ---------------- warp + per-channel dot: sim[b,d,h,w]
__global__ void k_warp_sim(const float* __restrict__ featT,
                           const float* __restrict__ depth_values,
                           const float* __restrict__ projws,
                           int view, float* __restrict__ simbuf) {
  int idx = blockIdx.x * 256 + threadIdx.x;
  if (idx >= Bn * Dn * HWn) return;
  int w = idx % Wn;
  int h = (idx / Wn) % Hn;
  int b = idx / (Dn * HWn);
  const float* pj = projws + ((view - 1) * Bn + b) * 12;
  float depth = depth_values[idx];
  float fx = (float)w, fy = (float)h;
  float px = (pj[0]*fx + pj[1]*fy + pj[2]) * depth + pj[9];
  float py = (pj[3]*fx + pj[4]*fy + pj[5]) * depth + pj[10];
  float pz = (pj[6]*fx + pj[7]*fy + pj[8]) * depth + pj[11];
  float iz = 1.f / pz;
  float x = px * iz, y = py * iz;          // (gx+1)*0.5*(W-1) round-trips to px/z
  float x0 = floorf(x), y0 = floorf(y);
  float wx = x - x0, wy = y - y0;

  const float4* refp = (const float4*)(featT + ((b * Hn + h) * Wn + w) * Cn); // v=0
  float4 rf[8];
  #pragma unroll
  for (int j = 0; j < 8; ++j) rf[j] = refp[j];

  const float* srcbase = featT + (size_t)(view * Bn + b) * Hn * Wn * Cn;
  float sim = 0.f;
  #pragma unroll
  for (int t = 0; t < 4; ++t) {
    float xf = x0 + (float)(t & 1);
    float yf = y0 + (float)(t >> 1);
    float wt = ((t & 1) ? wx : 1.f - wx) * ((t >> 1) ? wy : 1.f - wy);
    float val = ((xf >= 0.f) && (xf <= (float)(Wn - 1)) &&
                 (yf >= 0.f) && (yf <= (float)(Hn - 1))) ? 1.f : 0.f;
    float xc = fminf(fmaxf(xf, 0.f), (float)(Wn - 1));
    float yc = fminf(fmaxf(yf, 0.f), (float)(Hn - 1));
    int xi = (int)xc, yi = (int)yc;
    const float4* sp = (const float4*)(srcbase + (yi * Wn + xi) * Cn);
    float dot = 0.f;
    #pragma unroll
    for (int j = 0; j < 8; ++j) {
      float4 sv = sp[j];
      dot += sv.x * rf[j].x + sv.y * rf[j].y + sv.z * rf[j].z + sv.w * rf[j].w;
    }
    sim += wt * val * dot;
  }
  simbuf[idx] = sim * (1.f / (float)Cn);
}

// ---------------- pixelwise net (WMMA for the 16->8 layer) + max over D + accumulate
// One wave handles 16 pixel positions; per-d: B frag = hidden activations (16 pos x 16 ch),
// A frag = w1 zero-padded to 16x32 f16, D (f32) gives w1@h0 for channels 0..7 in lanes 0..15.
__global__ void k_pixelwise(const float* __restrict__ simbuf,
                            const float* __restrict__ w0, const float* __restrict__ g0,
                            const float* __restrict__ b0, const float* __restrict__ m0,
                            const float* __restrict__ v0,
                            const float* __restrict__ w1, const float* __restrict__ g1,
                            const float* __restrict__ b1, const float* __restrict__ m1,
                            const float* __restrict__ v1,
                            const float* __restrict__ w2, const float* __restrict__ b2,
                            int view,
                            float* __restrict__ sim_sum, float* __restrict__ w_sum,
                            float* __restrict__ o_vw) {
  int lane = threadIdx.x & 31;
  int gw   = blockIdx.x * (blockDim.x >> 5) + (threadIdx.x >> 5);
  int base = gw * 16;
  int pos  = base + (lane & 15);   // lanes 16..31 mirror 0..15 (their B frag is zero pad)
  int b    = pos / HWn;
  int hw   = pos - b * HWn;
  bool lo  = lane < 16;

  // fold BN0 into affine: h0_k = relu(sim*A0[k] + B0[k])
  float A0[16], B0f[16];
  #pragma unroll
  for (int k = 0; k < 16; ++k) {
    float s = g0[k] * rsqrtf(v0[k] + 1e-5f);
    A0[k]  = w0[k] * s;
    B0f[k] = b0[k] - m0[k] * s;
  }
  float S1[8], T1[8], W2r[8];
  #pragma unroll
  for (int r = 0; r < 8; ++r) {
    float s = g1[r] * rsqrtf(v1[r] + 1e-5f);
    S1[r] = s; T1[r] = b1[r] - m1[r] * s; W2r[r] = w2[r];
  }
  float bias2 = b2[0];

  // A fragment: 16x32 f16 (rows 8..15 zero, K 16..31 zero)
  int  mrow = lane & 15;
  bool hi   = lane >= 16;
  v16h af;
  #pragma unroll
  for (int j = 0; j < 8; ++j) {
    float wv = w1[(mrow & 7) * 16 + j];
    float wv8 = w1[(mrow & 7) * 16 + 8 + j];
    af[j]     = (_Float16)((mrow < 8 && !hi) ? wv : 0.f);
    af[j + 8] = (_Float16)((mrow < 8 && !hi) ? wv8 : 0.f);
  }

  float vmax = 0.f;
  for (int d = 0; d < Dn; ++d) {
    float sv = simbuf[(b * Dn + d) * HWn + hw];
    v16h bf;
    #pragma unroll
    for (int k = 0; k < 16; ++k) {
      float h = fmaxf(sv * A0[k] + B0f[k], 0.f);
      bf[k] = lo ? (_Float16)h : (_Float16)0.f;   // lanes 16..31 carry the K=16..31 zero pad
    }
    v8f c = {};
    c = __builtin_amdgcn_wmma_f32_16x16x32_f16(false, af, false, bf,
                                               (short)0, c, false, false);
    float o = bias2;
    #pragma unroll
    for (int r = 0; r < 8; ++r)
      o += W2r[r] * fmaxf(c[r] * S1[r] + T1[r], 0.f);
    float sig = 1.f / (1.f + expf(-o));
    vmax = fmaxf(vmax, sig);
  }

  if (lo) {  // divergence only after all WMMAs
    o_vw[(b * (Vn - 1) + (view - 1)) * HWn + hw] = vmax;
    #pragma unroll
    for (int d = 0; d < Dn; ++d) {
      int idx = (b * Dn + d) * HWn + hw;
      sim_sum[idx] += simbuf[idx] * vmax;
    }
    w_sum[pos] += vmax;
  }
}

__global__ void k_similarity(const float* __restrict__ sim_sum,
                             const float* __restrict__ w_sum,
                             float* __restrict__ simil) {
  int i = blockIdx.x * 256 + threadIdx.x;
  if (i >= Bn * Dn * HWn) return;
  int hw = i % HWn;
  int b  = i / (Dn * HWn);
  simil[i] = sim_sum[i] / w_sum[b * HWn + hw];
}

__global__ void k_conv3d(const float* __restrict__ sm, const float* __restrict__ rw,
                         const float* __restrict__ rb, float* __restrict__ cost) {
  int i = blockIdx.x * 256 + threadIdx.x;
  if (i >= Bn * Dn * HWn) return;
  int w = i % Wn;
  int h = (i / Wn) % Hn;
  int d = (i / HWn) % Dn;
  int b = i / (Dn * HWn);
  float acc = 0.f;
  #pragma unroll
  for (int kd = -1; kd <= 1; ++kd)
    #pragma unroll
    for (int kh = -1; kh <= 1; ++kh)
      #pragma unroll
      for (int kw = -1; kw <= 1; ++kw) {
        int dd = d + kd, hh = h + kh, ww = w + kw;
        if (dd < 0 || dd >= Dn || hh < 0 || hh >= Hn || ww < 0 || ww >= Wn) continue;
        acc += rw[(kd + 1) * 9 + (kh + 1) * 3 + (kw + 1)] *
               sm[((b * Dn + dd) * Hn + hh) * Wn + ww];
      }
  cost[i] = acc + rb[0];
}

__global__ void k_softmax(const float* __restrict__ cost,
                          const float* __restrict__ depth_values,
                          float* __restrict__ o_depth, float* __restrict__ o_conf,
                          float* __restrict__ o_prob) {
  int p = blockIdx.x * 256 + threadIdx.x;
  if (p >= Bn * HWn) return;
  int b = p / HWn;
  int hw = p - b * HWn;
  float x[Dn];
  float mx = -3.4e38f, bv = -3.4e38f;
  int bi = 0;
  #pragma unroll
  for (int d = 0; d < Dn; ++d) {
    x[d] = cost[(b * Dn + d) * HWn + hw];
    mx = fmaxf(mx, x[d]);
    if (x[d] > bv) { bv = x[d]; bi = d; }   // first-occurrence argmax
  }
  float s = 0.f;
  #pragma unroll
  for (int d = 0; d < Dn; ++d) { x[d] = expf(x[d] - mx); s += x[d]; }
  float inv = 1.f / s;
  #pragma unroll
  for (int d = 0; d < Dn; ++d) o_prob[(b * Dn + d) * HWn + hw] = x[d] * inv;
  o_depth[p] = depth_values[(b * Dn + bi) * HWn + hw];
  o_conf[p]  = inv;   // exp(max-max)/sum == 1/sum
}

extern "C" void kernel_launch(void* const* d_in, const int* in_sizes, int n_in,
                              void* d_out, int out_size, void* d_ws, size_t ws_size,
                              hipStream_t stream) {
  const float* features     = (const float*)d_in[0];
  const float* pm           = (const float*)d_in[1];
  const float* depth_values = (const float*)d_in[2];
  // d_in[3] = num_depth (hardcoded Dn=32)
  const float* w0 = (const float*)d_in[4];
  const float* g0 = (const float*)d_in[5];
  const float* b0 = (const float*)d_in[6];
  const float* m0 = (const float*)d_in[7];
  const float* v0 = (const float*)d_in[8];
  const float* w1 = (const float*)d_in[9];
  const float* g1 = (const float*)d_in[10];
  const float* b1 = (const float*)d_in[11];
  const float* m1 = (const float*)d_in[12];
  const float* v1 = (const float*)d_in[13];
  const float* w2 = (const float*)d_in[14];
  const float* b2 = (const float*)d_in[15];
  const float* rw = (const float*)d_in[16];
  const float* rb = (const float*)d_in[17];

  float* out = (float*)d_out;
  float* o_depth = out;                          // 40960
  float* o_conf  = out + 40960;                  // 40960
  float* o_prob  = out + 81920;                  // 1310720
  float* o_cost  = out + 81920 + 1310720;        // 1310720
  float* o_vw    = o_cost + 1310720;             // 163840

  float* ws      = (float*)d_ws;
  float* featT   = ws;                           // 6,553,600
  float* simbuf  = featT + 6553600;              // 1,310,720
  float* sim_sum = simbuf + 1310720;             // 1,310,720
  float* w_sum   = sim_sum + 1310720;            // 40,960
  float* simil   = w_sum + 40960;                // 1,310,720
  float* projws  = simil + 1310720;              // 96

  const int NE = Bn * Dn * HWn;                  // 1,310,720

  k_transpose<<<(Vn * Bn * Cn * HWn + 255) / 256, 256, 0, stream>>>(features, featT);
  k_proj<<<1, 1, 0, stream>>>(pm, projws);
  k_init<<<(NE + 255) / 256, 256, 0, stream>>>(sim_sum, w_sum);

  for (int v = 1; v < Vn; ++v) {
    k_warp_sim<<<(NE + 255) / 256, 256, 0, stream>>>(featT, depth_values, projws, v, simbuf);
    // 2560 waves * 16 positions = 40960 = Bn*HWn exactly; 8 waves per block
    k_pixelwise<<<320, 256, 0, stream>>>(simbuf, w0, g0, b0, m0, v0,
                                         w1, g1, b1, m1, v1, w2, b2,
                                         v, sim_sum, w_sum, o_vw);
  }

  k_similarity<<<(NE + 255) / 256, 256, 0, stream>>>(sim_sum, w_sum, simil);
  k_conv3d<<<(NE + 255) / 256, 256, 0, stream>>>(simil, rw, rb, o_cost);
  k_softmax<<<(Bn * HWn + 255) / 256, 256, 0, stream>>>(o_cost, depth_values,
                                                        o_depth, o_conf, o_prob);
}
